// ChemProp_15745350107778
// MI455X (gfx1250) — matrix operand order repaired and checked by reading, not compiled, not run on hardware
//
#include <hip/hip_runtime.h>

#define HID      256
#define NODE_DIM 128
#define EDGE_DIM 64

typedef __attribute__((ext_vector_type(16))) __bf16 v16bf;
typedef __attribute__((ext_vector_type(8)))  __bf16 v8bf;
typedef __attribute__((ext_vector_type(4)))  __bf16 v4bf;
typedef __attribute__((ext_vector_type(8)))  float  v8f;
typedef __attribute__((ext_vector_type(4)))  float  v4f;

union BFrag { v16bf v; v8bf h[2]; };

__device__ __forceinline__ v8f wmma_bf16(v16bf a, v16bf b, v8f c) {
    // D = A(16x32 bf16) x B(32x16 bf16) + C(16x16 f32)
    return __builtin_amdgcn_wmma_f32_16x16x32_bf16(false, a, false, b, (short)0, c, false, false);
}

__device__ __forceinline__ void st_bf4(__bf16* p, float a, float b, float c, float d) {
    v4bf q = { (__bf16)a, (__bf16)b, (__bf16)c, (__bf16)d };
    *(v4bf*)p = q;                      // ds_store_b64
}

__device__ __forceinline__ v4f ld4(const float* p) {
    return *(const v4f*)p;
}

__device__ __forceinline__ v4f ld_nt4(const float* p) {
    return __builtin_nontemporal_load((const v4f*)p);   // TH=NT: keep L2 for node_sum
}

// ---- GEMM core: 64 rows (LDS bf16 A tile, 4 x 16-row M tiles) x 256 cols ----
// Wave w owns output col tiles [32w..32w+15] and [32w+16..32w+31] for ALL 4 M tiles,
// so each B fragment (weights, from L2) is reused 4x.
template<int K, int LDA, int MT>
__device__ __forceinline__ void tile_gemm(const __bf16* As, const __bf16* __restrict__ Wt,
                                          v8f (&acc)[MT][2]) {
    const int lane = threadIdx.x & 31;
    const int wave = threadIdx.x >> 5;
    const int rc   = lane & 15;   // A row / B col within a 16x16 tile
    const int half = lane >> 4;
    const __bf16* Abase = As + rc * LDA + half * 8;
    const __bf16* B0    = Wt + (size_t)(wave * 32 + rc) * K + half * 8;
    const __bf16* B1    = B0 + (size_t)16 * K;
#pragma unroll
    for (int k = 0; k < K; k += 32) {
        BFrag b0, b1;
        b0.h[0] = *(const v8bf*)(B0 + k);          // K = k + 8h + 0..7
        b0.h[1] = *(const v8bf*)(B0 + k + 16);     // K = k + 16 + 8h + 0..7
        b1.h[0] = *(const v8bf*)(B1 + k);
        b1.h[1] = *(const v8bf*)(B1 + k + 16);
#pragma unroll
        for (int mt = 0; mt < MT; ++mt) {
            BFrag a;
            const __bf16* Ar = Abase + mt * 16 * LDA;
            a.h[0] = *(const v8bf*)(Ar + k);        // K = k + 8h + 0..7
            a.h[1] = *(const v8bf*)(Ar + k + 16);   // K = k + 16 + 8h + 0..7
            acc[mt][0] = wmma_bf16(a.v, b0.v, acc[mt][0]);
            acc[mt][1] = wmma_bf16(a.v, b1.v, acc[mt][1]);
        }
    }
}

// ---- utility kernels ----
__global__ void k_zero(float* __restrict__ p, size_t n) {
    size_t i = (size_t)blockIdx.x * blockDim.x + threadIdx.x;
    if (i < n) p[i] = 0.0f;
}

// Wt[n*K + k] = bf16(W[k*256 + n])   (transpose so B fragments are contiguous-K loads)
__global__ void k_wT(const float* __restrict__ W, __bf16* __restrict__ Wt, int K) {
    int i = blockIdx.x * blockDim.x + threadIdx.x;
    if (i >= K * HID) return;
    int n = i / K, k = i - n * K;
    Wt[i] = (__bf16)W[(size_t)k * HID + n];
}

// ---- GEMM 1: h0 = relu([x[src] | edge_attr] @ W1); scatter ns += h0 at dst ----
__global__ __launch_bounds__(256) void k_msg_init(
    const float* __restrict__ x, const float* __restrict__ eattr,
    const int* __restrict__ src, const int* __restrict__ dst,
    const __bf16* __restrict__ Wt1, float* __restrict__ h0,
    float* __restrict__ ns, int E)
{
    constexpr int K = NODE_DIM + EDGE_DIM;  // 192
    constexpr int LDA = K + 8;              // 200
    __shared__ alignas(16) __bf16 As[64 * LDA];

    const int e0 = blockIdx.x * 64;
    {
        const int t = threadIdx.x;
        const int r = t >> 2;               // 0..63, 4 threads per row
        const int c4 = (t & 3) * 4;
        const int e = min(e0 + r, E - 1);
        const float* xr = x + (size_t)src[e] * NODE_DIM;
        const float* er = eattr + (size_t)e * EDGE_DIM;
        __bf16* Ar = As + r * LDA;
#pragma unroll
        for (int j = 0; j < K / 16; ++j) {
            const int cc = c4 + 16 * j;     // cc<128 uniform per j
            v4f v = (cc < NODE_DIM) ? ld4(xr + cc)
                                    : ld_nt4(er + cc - NODE_DIM);
            st_bf4(Ar + cc, v.x, v.y, v.z, v.w);
        }
    }
    __syncthreads();

    v8f acc[4][2];
#pragma unroll
    for (int mt = 0; mt < 4; ++mt) { acc[mt][0] = {}; acc[mt][1] = {}; }
    tile_gemm<K, LDA, 4>(As, Wt1, acc);

    const int lane = threadIdx.x & 31;
    const int wave = threadIdx.x >> 5;
    const int half = lane >> 4;
    const int n0 = wave * 32 + (lane & 15);
    const int n1 = n0 + 16;
#pragma unroll
    for (int mt = 0; mt < 4; ++mt) {
#pragma unroll
        for (int v = 0; v < 8; ++v) {
            const int e = e0 + mt * 16 + v + 8 * half;
            if (e < E) {
                float f0 = fmaxf(acc[mt][0][v], 0.0f);
                float f1 = fmaxf(acc[mt][1][v], 0.0f);
                __builtin_nontemporal_store(f0, h0 + (size_t)e * HID + n0);
                __builtin_nontemporal_store(f1, h0 + (size_t)e * HID + n1);
                const size_t db = (size_t)dst[e] * HID;
                atomicAdd(ns + db + n0, f0);
                atomicAdd(ns + db + n1, f1);
            }
        }
    }
}

// ---- GEMM 2 (message step): m = ns_in[src] - h_prev[rev];
//      h_new = relu(h0 + m @ W2); optional store; scatter ns_out += h_new at dst ----
__global__ __launch_bounds__(256) void k_msg_step(
    const float* __restrict__ h0b, const float* __restrict__ hprev,
    const float* __restrict__ ns_in,
    const int* __restrict__ src, const int* __restrict__ dst, const int* __restrict__ rev,
    const __bf16* __restrict__ Wt2, float* __restrict__ h_out,
    float* __restrict__ ns_out, int E)
{
    constexpr int K = HID;       // 256
    constexpr int LDA = K + 8;   // 264
    __shared__ alignas(16) __bf16 As[64 * LDA];

    const int e0 = blockIdx.x * 64;
    {
        const int t = threadIdx.x;
        const int r = t >> 2;
        const int c4 = (t & 3) * 4;
        const int e = min(e0 + r, E - 1);
        const float* nsr = ns_in + (size_t)src[e] * HID;   // L2-resident
        const float* hr  = hprev + (size_t)rev[e] * HID;   // single-use stream
        __bf16* Ar = As + r * LDA;
#pragma unroll
        for (int j = 0; j < K / 16; ++j) {
            const int cc = c4 + 16 * j;
            v4f a = ld4(nsr + cc);
            v4f b = ld_nt4(hr + cc);
            st_bf4(Ar + cc, a.x - b.x, a.y - b.y, a.z - b.z, a.w - b.w);
        }
    }
    __syncthreads();

    v8f acc[4][2];
#pragma unroll
    for (int mt = 0; mt < 4; ++mt) { acc[mt][0] = {}; acc[mt][1] = {}; }
    tile_gemm<K, LDA, 4>(As, Wt2, acc);

    const int lane = threadIdx.x & 31;
    const int wave = threadIdx.x >> 5;
    const int half = lane >> 4;
    const int n0 = wave * 32 + (lane & 15);
    const int n1 = n0 + 16;
#pragma unroll
    for (int mt = 0; mt < 4; ++mt) {
#pragma unroll
        for (int v = 0; v < 8; ++v) {
            const int e = e0 + mt * 16 + v + 8 * half;
            if (e < E) {
                const size_t eb = (size_t)e * HID;
                float f0 = fmaxf(__builtin_nontemporal_load(h0b + eb + n0) + acc[mt][0][v], 0.0f);
                float f1 = fmaxf(__builtin_nontemporal_load(h0b + eb + n1) + acc[mt][1][v], 0.0f);
                if (h_out) {
                    __builtin_nontemporal_store(f0, h_out + eb + n0);
                    __builtin_nontemporal_store(f1, h_out + eb + n1);
                }
                const size_t db = (size_t)dst[e] * HID;
                atomicAdd(ns_out + db + n0, f0);
                atomicAdd(ns_out + db + n1, f1);
            }
        }
    }
}

// ---- GEMM 3 (readout): na = relu([x | v_msg] @ W3 + b3); out[batch] += na ----
__global__ __launch_bounds__(256) void k_readout(
    const float* __restrict__ x, const float* __restrict__ vmsg,
    const int* __restrict__ batch, const __bf16* __restrict__ Wt3,
    const float* __restrict__ b3, float* __restrict__ out, int N)
{
    constexpr int K = NODE_DIM + HID;  // 384
    constexpr int LDA = K + 8;         // 392
    __shared__ alignas(16) __bf16 As[64 * LDA];

    const int v0n = blockIdx.x * 64;
    {
        const int t = threadIdx.x;
        const int r = t >> 2;
        const int c4 = (t & 3) * 4;
        const int nd = min(v0n + r, N - 1);
        const float* xr = x + (size_t)nd * NODE_DIM;
        const float* vm = vmsg + (size_t)nd * HID;
        __bf16* Ar = As + r * LDA;
#pragma unroll
        for (int j = 0; j < K / 16; ++j) {
            const int cc = c4 + 16 * j;
            v4f v = (cc < NODE_DIM) ? ld4(xr + cc)
                                    : ld4(vm + cc - NODE_DIM);
            st_bf4(Ar + cc, v.x, v.y, v.z, v.w);
        }
    }
    __syncthreads();

    v8f acc[4][2];
#pragma unroll
    for (int mt = 0; mt < 4; ++mt) { acc[mt][0] = {}; acc[mt][1] = {}; }
    tile_gemm<K, LDA, 4>(As, Wt3, acc);

    const int lane = threadIdx.x & 31;
    const int wave = threadIdx.x >> 5;
    const int half = lane >> 4;
    const int n0 = wave * 32 + (lane & 15);
    const int n1 = n0 + 16;
    const float bias0 = b3[n0], bias1 = b3[n1];
#pragma unroll
    for (int mt = 0; mt < 4; ++mt) {
#pragma unroll
        for (int v = 0; v < 8; ++v) {
            const int nd = v0n + mt * 16 + v + 8 * half;
            if (nd < N) {
                float f0 = fmaxf(acc[mt][0][v] + bias0, 0.0f);
                float f1 = fmaxf(acc[mt][1][v] + bias1, 0.0f);
                const size_t gb = (size_t)batch[nd] * HID;
                atomicAdd(out + gb + n0, f0);
                atomicAdd(out + gb + n1, f1);
            }
        }
    }
}

extern "C" void kernel_launch(void* const* d_in, const int* in_sizes, int n_in,
                              void* d_out, int out_size, void* d_ws, size_t ws_size,
                              hipStream_t stream) {
    const float* x     = (const float*)d_in[0];
    const int*   eidx  = (const int*)d_in[1];   // (2,E): src = eidx, dst = eidx+E
    const int*   rev   = (const int*)d_in[2];
    const float* eattr = (const float*)d_in[3];
    const int*   batch = (const int*)d_in[5];
    const float* W1    = (const float*)d_in[6];
    const float* W2    = (const float*)d_in[7];
    const float* W3    = (const float*)d_in[8];
    const float* b3    = (const float*)d_in[9];
    float* out = (float*)d_out;

    const int E = in_sizes[2];
    const int N = in_sizes[0] / NODE_DIM;
    const int G = out_size / HID;
    const int* src = eidx;
    const int* dst = eidx + E;

    auto au = [](size_t v) { return (v + 255) & ~(size_t)255; };
    char* p = (char*)d_ws;
    __bf16* Wt1 = (__bf16*)p; p += au((size_t)192 * HID * sizeof(__bf16));
    __bf16* Wt2 = (__bf16*)p; p += au((size_t)256 * HID * sizeof(__bf16));
    __bf16* Wt3 = (__bf16*)p; p += au((size_t)384 * HID * sizeof(__bf16));
    float*  h0  = (float*)p;  p += au((size_t)E * HID * sizeof(float));
    float*  h1  = (float*)p;  p += au((size_t)E * HID * sizeof(float));
    float*  ns0 = (float*)p;  p += au((size_t)N * HID * sizeof(float));
    float*  ns1 = (float*)p;

    const size_t nsE  = (size_t)N * HID;
    const size_t outE = (size_t)G * HID;
    const int eb = (E + 63) / 64;
    const int nb = (N + 63) / 64;

    k_wT<<<(192 * HID + 255) / 256, 256, 0, stream>>>(W1, Wt1, 192);
    k_wT<<<(256 * HID + 255) / 256, 256, 0, stream>>>(W2, Wt2, 256);
    k_wT<<<(384 * HID + 255) / 256, 256, 0, stream>>>(W3, Wt3, 384);

    k_zero<<<(int)((nsE + 255) / 256), 256, 0, stream>>>(ns0, nsE);
    k_msg_init<<<eb, 256, 0, stream>>>(x, eattr, src, dst, Wt1, h0, ns0, E);

    // depth step 1: h1 = relu(h0 + (ns0[src] - h0[rev]) @ W2); build ns1
    k_zero<<<(int)((nsE + 255) / 256), 256, 0, stream>>>(ns1, nsE);
    k_msg_step<<<eb, 256, 0, stream>>>(h0, h0, ns0, src, dst, rev, Wt2, h1, ns1, E);

    // depth step 2: only the scatter of h2 is needed downstream -> ns0 reused as ns2
    k_zero<<<(int)((nsE + 255) / 256), 256, 0, stream>>>(ns0, nsE);
    k_msg_step<<<eb, 256, 0, stream>>>(h0, h1, ns1, src, dst, rev, Wt2, nullptr, ns0, E);

    // readout + global add pool
    k_zero<<<(int)((outE + 255) / 256), 256, 0, stream>>>(out, outE);
    k_readout<<<nb, 256, 0, stream>>>(x, ns0, batch, Wt3, b3, out, N);
}
